// DeepPriorDeformer_57372173140352
// MI455X (gfx1250) — matrix operand order repaired and checked by reading, not compiled, not run on hardware
//
#include <hip/hip_runtime.h>

// DeepPriorDeformer forward for MI455X (gfx1250, wave32, WMMA).
//  - every matmul -> one tiled WMMA GEMM (v_wmma_f32_16x16x32_f16, f32 acc)
//  - all GEMM operands are f16 in memory (converted once; producers dual-write
//    f32 + f16 shadows), halving GEMM-side HBM reads
//  - tiles staged global->LDS with CDNA5 async copies
//    (global_load_async_to_lds_b128 + s_wait_asynccnt)
//  - NO guarded slow path: partial tiles use clamped (always in-bounds)
//    staging; out-of-range rows/cols are discarded in the epilogue
//  - B tile kept N-major in LDS so every WMMA fragment is ds_load_b128 pairs
//  - eval-BatchNorm folded to per-channel scale/shift in the GEMM epilogue
//  - index_select exploited: only the selected class rows of am[2]/dm2[2]
//    are ever computed (weight-row offset per batch)

typedef __attribute__((ext_vector_type(16))) _Float16 v16h;
typedef __attribute__((ext_vector_type(8)))  _Float16 v8h;
typedef __attribute__((ext_vector_type(8)))  float    v8f;
typedef long long ll;
typedef __attribute__((address_space(3))) char lds_char;

constexpr int BSZ = 16, NPT = 1024, NPR = 1024, NCLS = 6, EDIM = 128, NH = 4, HDIM = 32;
constexpr float ATT_SCALE = 0.17677669529663687f;   // 1/sqrt(32)
constexpr float LEPS = 1e-5f;

// ---------------- workspace layout (BYTE offsets) -------------------
constexpr ll B_SCORES   = 0;                           // f32 scores, 33.5MB max
constexpr ll B_AFSEL    = B_SCORES   + 33554432LL;     // f32 (B,P,N) 67MB; phase-B alias: f16 MHA temps
constexpr ll B_SOFTAF   = B_AFSEL    + 67108864LL;     // f32 (B,N,P) 67MB
constexpr ll B_QPRIORF  = B_SOFTAF   + 67108864LL;     // f32 (B,128,P)
constexpr ll B_QPTSF    = B_QPRIORF  + 8388608LL;      // f32 (B,128,128)
constexpr ll B_PFF      = B_QPTSF    + 1048576LL;      // f32 prior_feat
constexpr ll B_X3F      = B_PFF      + 8388608LL;
constexpr ll B_ATTNF    = B_X3F      + 8388608LL;      // f32 MHA out-proj (max Lq)
constexpr ll B_LOWRF    = B_ATTNF    + 8388608LL;      // f32 lowrank pre-softmax
constexpr ll B_QV3F     = B_LOWRF    + 8388608LL;      // f32 (B,3,P)
constexpr ll B_RGBG     = B_QV3F     + 196608LL;
constexpr ll B_PTSG     = B_RGBG     + 8192LL;
constexpr ll B_PRIORG   = B_PTSG     + 8192LL;
constexpr ll B_FOLD     = B_PRIORG   + 8192LL;         // 1536 f32
constexpr ll B_ROFF     = B_FOLD     + 6144LL;         // 32 ints
// ---- f16 arena ----
constexpr ll B_W16      = B_ROFF     + 256LL;          // all converted weights (~5.5MB)
constexpr ll B_PTSL16   = B_W16      + 6291456LL;
constexpr ll B_PRIORL16 = B_PTSL16   + 4194304LL;
constexpr ll B_DEF16    = B_PRIORL16 + 4194304LL;      // alias: refcat16, attin16
constexpr ll B_X1_16    = B_DEF16    + 12582912LL;     // alias: h1_16, a1_16
constexpr ll B_X2_16    = B_X1_16    + 12582912LL;     // alias: lowrank16, a2_16
constexpr ll B_WPF16    = B_X2_16    + 8388608LL;
constexpr ll B_QPRIOR16 = B_WPF16    + 524288LL;
constexpr ll B_KPTS16   = B_QPRIOR16 + 4194304LL;
constexpr ll B_QPTS16   = B_KPTS16   + 4194304LL;
constexpr ll B_PF16     = B_QPTS16   + 524288LL;
constexpr ll B_SOFTA16  = B_PF16     + 4194304LL;
constexpr ll B_QV1_16   = B_SOFTA16  + 33554432LL;
constexpr ll B_QV2_16   = B_QV1_16   + 2097152LL;      // end ~ 310.6MB
// phase-B f16 MHA temps live inside the (then unused) AFSEL block
constexpr ll B_QP16  = B_AFSEL;
constexpr ll B_KK16  = B_QP16  + 4194304LL;
constexpr ll B_VV16  = B_KK16  + 4194304LL;
constexpr ll B_SC16  = B_VV16  + 4194304LL;            // 16.8MB
constexpr ll B_CTX16 = B_SC16  + 16777216LL;           // total 33.5MB <= 67MB
constexpr ll B_REFCAT16 = B_DEF16;
constexpr ll B_H1_16    = B_X1_16;
constexpr ll B_LOWR16   = B_X2_16;

// ---------------- output layout (float offsets) ---------------------
constexpr ll O_OUT_ATT = 0;
constexpr ll O_OUT_NP  = O_OUT_ATT + (ll)BSZ * NPT * NPR;
constexpr ll O_OUT_QV  = O_OUT_NP  + (ll)BSZ * EDIM * NPT;
constexpr ll O_OUT_QO  = O_OUT_QV  + (ll)BSZ * NPR * 3;

// ======================= async global->LDS copy =====================
#define USE_ASYNC_G2L 1

__device__ __forceinline__ unsigned lds_addr(const void* p) {
  return (unsigned)(unsigned long long)(const lds_char*)p;   // flat -> LDS offset
}

__device__ __forceinline__ void g2l_b128(const _Float16* __restrict__ g,
                                         _Float16* __restrict__ l) {
#if USE_ASYNC_G2L
  asm volatile("global_load_async_to_lds_b128 %0, %1, off"
               :: "v"(lds_addr(l)), "v"((unsigned long long)(size_t)g)
               : "memory");
#else
  *(v8h*)l = *(const v8h*)g;
#endif
}
__device__ __forceinline__ void g2l_fence() {
#if USE_ASYNC_G2L
  asm volatile("s_wait_asynccnt 0" ::: "memory");
#endif
}

// ======================= WMMA GEMM ==================================
// C[M,N](f32,opt) / C16[M,N](f16,opt) = act(alpha*A*B*scale[m]+shift[m])
//   AT=0: A (M,K) f16 row-major; AT=1: A stored (K,M)
//   BT=0: B (K,N) f16 row-major; BT=1: B stored (N,K)
//   rowOff: per-outer-batch weight-row offset (class select)
// Requirements (all satisfied by every call site):
//   K % 32 == 0 ; lda,ldb % 8 == 0 ; N % 8 == 0 ; (AT: M % 8 == 0)
#define BLK_M 64
#define BLK_N 64
#define BLK_K 32
#define LDS_S 40   // LDS row stride in halves (both tiles)

template <bool AT, bool BT, bool RELU>
__global__ __launch_bounds__(256) void gemm16_kernel(
    const _Float16* __restrict__ A, const _Float16* __restrict__ Bm,
    float* __restrict__ C, _Float16* __restrict__ C16,
    int M, int N, int K, int lda, int ldb, int ldc,
    ll sAb, ll sBb, ll sCb, int H2, ll sAh, ll sBh, ll sCh,
    const float* __restrict__ scaleVec, const float* __restrict__ shiftVec,
    float alpha, const int* __restrict__ rowOff) {
  __shared__ alignas(16) _Float16 lA[BLK_M * LDS_S];          // m-major [64][40]
  __shared__ alignas(16) _Float16 lB[BLK_N * LDS_S];          // n-major [64][40]

  const int z = blockIdx.z, bo = z / H2, ho = z - bo * H2;
  int roff = 0;
  A  += (ll)bo * sAb + (ll)ho * sAh;
  Bm += (ll)bo * sBb + (ll)ho * sBh;
  const ll cOff = (ll)bo * sCb + (ll)ho * sCh;
  if (!AT && rowOff) { roff = rowOff[bo]; A += (ll)roff * lda; }

  const int m0 = blockIdx.y * BLK_M, n0 = blockIdx.x * BLK_N;
  const int tid = threadIdx.x, lane = tid & 31, w = tid >> 5;
  const int wm = w >> 1, wn = w & 1;

  // clamped (always in-bounds) staging coordinates; excess rows/cols are
  // duplicates whose results are discarded in the epilogue
  const int aRow  = min(m0 + (tid >> 2), M - 1);          // A !AT: row, chunk (tid&3)*8
  const int aColC = min(m0 + (tid & 7) * 8, M - 8);       // A AT: col chunk base
  const int bRow  = min(n0 + (tid >> 2), N - 1);          // B BT: row, chunk (tid&3)*8
  const int bColC = min(n0 + (tid & 7) * 8, N - 8);       // B !BT: col chunk base

  v8f acc0 = {0.f, 0.f, 0.f, 0.f, 0.f, 0.f, 0.f, 0.f};
  v8f acc1 = {0.f, 0.f, 0.f, 0.f, 0.f, 0.f, 0.f, 0.f};

  for (int k0 = 0; k0 < K; k0 += BLK_K) {
    // ---- stage A ------------------------------------------------------
    if (!AT) {
      const int m = tid >> 2, k = (tid & 3) * 8;
      g2l_b128(&A[(ll)aRow * lda + k0 + k], &lA[m * LDS_S + k]);
    } else {
      const int k = tid >> 3, mc = (tid & 7) * 8;
      v8h v = *(const v8h*)&A[(ll)(k0 + k) * lda + aColC];
      #pragma unroll
      for (int j = 0; j < 8; ++j) lA[(mc + j) * LDS_S + k] = v[j];
    }
    // ---- stage B (always N-major in LDS) ------------------------------
    if (BT) {
      const int n = tid >> 2, k = (tid & 3) * 8;
      g2l_b128(&Bm[(ll)bRow * ldb + k0 + k], &lB[n * LDS_S + k]);
    } else {
      const int k = tid >> 3, nc = (tid & 7) * 8;
      v8h v = *(const v8h*)&Bm[(ll)(k0 + k) * ldb + bColC];
      #pragma unroll
      for (int j = 0; j < 8; ++j) lB[(nc + j) * LDS_S + k] = v[j];
    }
    g2l_fence();
    __syncthreads();

    // ---- A fragment: lane = M row; K = (lane>=16)*8 + {0..7}, +16 -------
    const int ar = wm * 16 + (lane & 15);
    const int akb = (lane >> 4) * 8;
    v8h alo = *(const v8h*)&lA[ar * LDS_S + akb];
    v8h ahi = *(const v8h*)&lA[ar * LDS_S + akb + 16];
    v16h afr;
    #pragma unroll
    for (int j = 0; j < 8; ++j) { afr[j] = alo[j]; afr[j + 8] = ahi[j]; }

    // ---- B fragments: lane = N col; K = (lane>=16)*8 + {0..7}, +16 -------
    const int bkb = (lane >> 4) * 8;
    int bc = wn * 32 + (lane & 15);
    v8h lo = *(const v8h*)&lB[bc * LDS_S + bkb];
    v8h hi = *(const v8h*)&lB[bc * LDS_S + bkb + 16];
    v16h bfr0, bfr1;
    #pragma unroll
    for (int j = 0; j < 8; ++j) { bfr0[j] = lo[j]; bfr0[j + 8] = hi[j]; }
    bc += 16;
    lo = *(const v8h*)&lB[bc * LDS_S + bkb];
    hi = *(const v8h*)&lB[bc * LDS_S + bkb + 16];
    #pragma unroll
    for (int j = 0; j < 8; ++j) { bfr1[j] = lo[j]; bfr1[j + 8] = hi[j]; }

    acc0 = __builtin_amdgcn_wmma_f32_16x16x32_f16(false, afr, false, bfr0,
                                                  (short)0, acc0, false, false);
    acc1 = __builtin_amdgcn_wmma_f32_16x16x32_f16(false, afr, false, bfr1,
                                                  (short)0, acc1, false, false);
    __syncthreads();
  }

  // ---- epilogue -------------------------------------------------------
  float* Cb = C ? C + cOff : nullptr;
  _Float16* C16b = C16 ? C16 + cOff : nullptr;
  #pragma unroll
  for (int r = 0; r < 8; ++r) {
    const int ml = wm * 16 + r + ((lane >> 4) << 3);
    const int gm = m0 + ml;
    if (gm >= M) continue;
    const float sc = scaleVec ? scaleVec[roff + gm] : 1.f;
    const float sh = shiftVec ? shiftVec[roff + gm] : 0.f;
    int gn = n0 + wn * 32 + (lane & 15);
    if (gn < N) {
      float v = acc0[r] * alpha * sc + sh;
      if (RELU) v = fmaxf(v, 0.f);
      if (Cb)   Cb[(ll)gm * ldc + gn] = v;
      if (C16b) C16b[(ll)gm * ldc + gn] = (_Float16)v;
    }
    gn += 16;
    if (gn < N) {
      float v = acc1[r] * alpha * sc + sh;
      if (RELU) v = fmaxf(v, 0.f);
      if (Cb)   Cb[(ll)gm * ldc + gn] = v;
      if (C16b) C16b[(ll)gm * ldc + gn] = (_Float16)v;
    }
  }
}

// ======================= support kernels ============================
__global__ void cvt_f16_kernel(const float* __restrict__ in, _Float16* __restrict__ out, ll n) {
  const ll i = (ll)blockIdx.x * 256 + threadIdx.x;
  if (i < n) out[i] = (_Float16)in[i];
}

__global__ void row_softmax_kernel(const float* __restrict__ in, float* __restrict__ outF,
                                   _Float16* __restrict__ out16, int L) {
  __shared__ float red[256];
  const ll row = blockIdx.x;
  const float* x = in + row * (ll)L;
  const int t = threadIdx.x;
  float mx = -3.4e38f;
  for (int i = t; i < L; i += 256) mx = fmaxf(mx, x[i]);
  red[t] = mx; __syncthreads();
  for (int s = 128; s > 0; s >>= 1) { if (t < s) red[t] = fmaxf(red[t], red[t + s]); __syncthreads(); }
  mx = red[0]; __syncthreads();
  float sum = 0.f;
  for (int i = t; i < L; i += 256) sum += __expf(x[i] - mx);
  red[t] = sum; __syncthreads();
  for (int s = 128; s > 0; s >>= 1) { if (t < s) red[t] += red[t + s]; __syncthreads(); }
  const float inv = 1.f / red[0];
  for (int i = t; i < L; i += 256) {
    const float v = __expf(x[i] - mx) * inv;
    if (outF)  outF[row * (ll)L + i] = v;
    if (out16) out16[row * (ll)L + i] = (_Float16)v;
  }
}

__global__ void row_mean_kernel(const float* __restrict__ in, float* __restrict__ out, int L) {
  __shared__ float red[256];
  const ll row = blockIdx.x;
  const float* x = in + row * (ll)L;
  const int t = threadIdx.x;
  float s = 0.f;
  for (int i = t; i < L; i += 256) s += x[i];
  red[t] = s; __syncthreads();
  for (int st = 128; st > 0; st >>= 1) { if (t < st) red[t] += red[t + st]; __syncthreads(); }
  if (t == 0) out[row] = red[0] / (float)L;
}

// x = LN(x + att) over E channels; dual-writes f32 + f16 shadow
__global__ void ln_residual_kernel(float* __restrict__ x, _Float16* __restrict__ x16,
                                   const float* __restrict__ att,
                                   const float* __restrict__ g, const float* __restrict__ b,
                                   int L) {
  const ll i = (ll)blockIdx.x * blockDim.x + threadIdx.x;
  if (i >= (ll)BSZ * L) return;
  const int bb = (int)(i / L), l = (int)(i % L);
  const ll base = (ll)bb * EDIM * L + l;
  float s = 0.f, ss = 0.f;
  for (int e = 0; e < EDIM; ++e) {
    const float v = x[base + (ll)e * L] + att[base + (ll)e * L];
    s += v; ss += v * v;
  }
  const float mu  = s * (1.f / EDIM);
  const float var = ss * (1.f / EDIM) - mu * mu;
  const float inv = rsqrtf(var + LEPS);
  for (int e = 0; e < EDIM; ++e) {
    const ll o = base + (ll)e * L;
    const float v = (x[o] + att[o] - mu) * inv * g[e] + b[e];
    x[o] = v;
    x16[o] = (_Float16)v;
  }
}

__global__ void concat2_kernel(const float* __restrict__ a, const float* __restrict__ b2,
                               _Float16* __restrict__ out) {  // -> f16 (B,256,N)
  const ll i = (ll)blockIdx.x * 256 + threadIdx.x;
  if (i >= (ll)BSZ * 256 * NPT) return;
  const int n = (int)(i % NPT); const ll t = i / NPT;
  const int c = (int)(t % 256); const int bb = (int)(t / 256);
  const float v = (c < 128) ? a[((ll)bb * 128 + c) * NPT + n]
                            : b2[((ll)bb * 128 + (c - 128)) * NPT + n];
  out[i] = (_Float16)v;
}

__global__ void deform_concat_kernel(const float* __restrict__ pf, const float* __restrict__ ptsg,
                                     const float* __restrict__ rgbg, _Float16* __restrict__ out) {
  const ll i = (ll)blockIdx.x * 256 + threadIdx.x;
  if (i >= (ll)BSZ * 384 * NPR) return;
  const int n = (int)(i % NPR); const ll t = i / NPR;
  const int c = (int)(t % 384); const int bb = (int)(t / 384);
  float v;
  if (c < 128)      v = pf[((ll)bb * 128 + c) * NPR + n];
  else if (c < 256) v = ptsg[bb * 128 + (c - 128)];
  else              v = rgbg[bb * 128 + (c - 256)];
  out[i] = (_Float16)v;
}

__global__ void attin_concat_kernel(const float* __restrict__ ptsl, const float* __restrict__ rgbl,
                                    const float* __restrict__ priorg, _Float16* __restrict__ out) {
  const ll i = (ll)blockIdx.x * 256 + threadIdx.x;
  if (i >= (ll)BSZ * 384 * NPT) return;
  const int n = (int)(i % NPT); const ll t = i / NPT;
  const int c = (int)(t % 384); const int bb = (int)(t / 384);
  float v;
  if (c < 128)      v = ptsl[((ll)bb * 128 + c) * NPT + n];
  else if (c < 256) v = rgbl[((ll)bb * 128 + (c - 128)) * NPT + n];
  else              v = priorg[bb * 128 + (c - 256)];
  out[i] = (_Float16)v;
}

__global__ void addrelu_kernel(float* __restrict__ a, _Float16* __restrict__ a16,
                               const float* __restrict__ b, ll total) {
  const ll i = (ll)blockIdx.x * 256 + threadIdx.x;
  if (i < total) {
    const float v = fmaxf(a[i] + b[i], 0.f);
    a[i] = v;
    a16[i] = (_Float16)v;
  }
}

__global__ void bnfold_kernel(const float* __restrict__ g, const float* __restrict__ b,
                              const float* __restrict__ m, const float* __restrict__ v,
                              const float* __restrict__ cb, float* __restrict__ scale,
                              float* __restrict__ shift, int C) {
  const int c = blockIdx.x * blockDim.x + threadIdx.x;
  if (c < C) {
    const float inv = g[c] * rsqrtf(v[c] + LEPS);
    scale[c] = inv;
    shift[c] = cb[c] * inv + b[c] - m[c] * inv;
  }
}

__global__ void rowoff_kernel(const int* __restrict__ index, int* __restrict__ roffP,
                              int* __restrict__ roff3) {
  const int b = threadIdx.x;
  if (b < BSZ) {
    const int cls = index[b] % NCLS;
    roffP[b] = cls * NPR;
    roff3[b] = cls * 3;
  }
}

__global__ void transpose_kernel(const float* __restrict__ in, float* __restrict__ out,
                                 int R, int Ccols) {  // per z: (R,C) -> (C,R)
  __shared__ float t[32][33];
  in  += (ll)blockIdx.z * R * Ccols;
  out += (ll)blockIdx.z * R * Ccols;
  const int r0 = blockIdx.y * 32, c0 = blockIdx.x * 32;
  const int tx = threadIdx.x & 31, ty = threadIdx.x >> 5;
  for (int j = ty; j < 32; j += 8) {
    const int r = r0 + j, c = c0 + tx;
    t[j][tx] = (r < R && c < Ccols) ? in[(ll)r * Ccols + c] : 0.f;
  }
  __syncthreads();
  for (int j = ty; j < 32; j += 8) {
    const int c = c0 + j, r = r0 + tx;
    if (c < Ccols && r < R) out[(ll)c * R + r] = t[tx][j];
  }
}

__global__ void qv_out_kernel(const float* __restrict__ qv3, float* __restrict__ out) {
  const ll i = (ll)blockIdx.x * 256 + threadIdx.x;  // over B*P
  if (i >= (ll)BSZ * NPR) return;
  const int b = (int)(i / NPR), p = (int)(i % NPR);
  #pragma unroll
  for (int d = 0; d < 3; ++d)
    out[i * 3 + d] = qv3[((ll)b * 3 + d) * NPR + p];
}

__global__ void qo_kernel(const float* __restrict__ Qv, const float* __restrict__ softA,
                          float* __restrict__ out) {
  const ll i = (ll)blockIdx.x * 256 + threadIdx.x;  // over B*N
  if (i >= (ll)BSZ * NPT) return;
  const int b = (int)(i / NPT), n = (int)(i % NPT);
  const float* sa = softA + ((ll)b * NPT + n) * NPR;
  const float* qv = Qv + (ll)b * NPR * 3;
  float s0 = 0.f, s1 = 0.f, s2 = 0.f;
  for (int p = 0; p < NPR; ++p) {
    const float w = sa[p];
    s0 += w * qv[p * 3 + 0];
    s1 += w * qv[p * 3 + 1];
    s2 += w * qv[p * 3 + 2];
  }
  out[i * 3 + 0] = s0; out[i * 3 + 1] = s1; out[i * 3 + 2] = s2;
}

// ======================= host helpers ===============================
static void gemm(hipStream_t s, int at, int bt, int relu,
                 const _Float16* A, const _Float16* B, float* C, _Float16* C16,
                 int M, int N, int K, int lda, int ldb, int ldc,
                 ll sAb, ll sBb, ll sCb, int H2, ll sAh, ll sBh, ll sCh,
                 int batch, const float* scale, const float* shift,
                 float alpha, const int* roff) {
  dim3 g((N + BLK_N - 1) / BLK_N, (M + BLK_M - 1) / BLK_M, batch * H2);
  dim3 b(256);
  if (at)
    gemm16_kernel<true, false, false><<<g, b, 0, s>>>(A, B, C, C16, M, N, K, lda, ldb, ldc,
        sAb, sBb, sCb, H2, sAh, sBh, sCh, scale, shift, alpha, roff);
  else if (bt)
    gemm16_kernel<false, true, false><<<g, b, 0, s>>>(A, B, C, C16, M, N, K, lda, ldb, ldc,
        sAb, sBb, sCb, H2, sAh, sBh, sCh, scale, shift, alpha, roff);
  else if (relu)
    gemm16_kernel<false, false, true><<<g, b, 0, s>>>(A, B, C, C16, M, N, K, lda, ldb, ldc,
        sAb, sBb, sCb, H2, sAh, sBh, sCh, scale, shift, alpha, roff);
  else
    gemm16_kernel<false, false, false><<<g, b, 0, s>>>(A, B, C, C16, M, N, K, lda, ldb, ldc,
        sAb, sBb, sCb, H2, sAh, sBh, sCh, scale, shift, alpha, roff);
}

struct MhaP {
  const float *in_b, *out_b, *ln_b, *ln_g;
  const _Float16 *in_w16, *out_w16;
};

// q (f32) + q16 (f16 shadow) updated in place: q = LN(q + MHA(q, kv, kv))
static void run_mha(hipStream_t s, char* wsb, float* q, _Float16* q16,
                    const _Float16* kv16, int Lq, int Lk, MhaP p) {
  _Float16* Qp16 = (_Float16*)(wsb + B_QP16);
  _Float16* Kk16 = (_Float16*)(wsb + B_KK16);
  _Float16* Vv16 = (_Float16*)(wsb + B_VV16);
  _Float16* Sc16 = (_Float16*)(wsb + B_SC16);
  _Float16* Cx16 = (_Float16*)(wsb + B_CTX16);
  float* ScF = (float*)(wsb + B_SCORES);
  float* AtF = (float*)(wsb + B_ATTNF);
  const int EE = EDIM * EDIM;
  // in-projections (f16-only outputs)
  gemm(s, 0, 0, 0, p.in_w16,          q16,  nullptr, Qp16, EDIM, Lq, EDIM,
       EDIM, Lq, Lq, 0, (ll)EDIM * Lq, (ll)EDIM * Lq, 1, 0, 0, 0, BSZ,
       nullptr, p.in_b, 1.f, nullptr);
  gemm(s, 0, 0, 0, p.in_w16 + EE,     kv16, nullptr, Kk16, EDIM, Lk, EDIM,
       EDIM, Lk, Lk, 0, (ll)EDIM * Lk, (ll)EDIM * Lk, 1, 0, 0, 0, BSZ,
       nullptr, p.in_b + EDIM, 1.f, nullptr);
  gemm(s, 0, 0, 0, p.in_w16 + 2 * EE, kv16, nullptr, Vv16, EDIM, Lk, EDIM,
       EDIM, Lk, Lk, 0, (ll)EDIM * Lk, (ll)EDIM * Lk, 1, 0, 0, 0, BSZ,
       nullptr, p.in_b + 2 * EDIM, 1.f, nullptr);
  // scores (Lq,Lk) = Qh^T Kh * SCALE, f32 out for the softmax
  gemm(s, 1, 0, 0, Qp16, Kk16, ScF, nullptr, Lq, Lk, HDIM, Lq, Lk, Lk,
       (ll)EDIM * Lq, (ll)EDIM * Lk, (ll)NH * Lq * Lk,
       NH, (ll)HDIM * Lq, (ll)HDIM * Lk, (ll)Lq * Lk,
       BSZ, nullptr, nullptr, ATT_SCALE, nullptr);
  row_softmax_kernel<<<BSZ * NH * Lq, 256, 0, s>>>(ScF, nullptr, Sc16, Lk);
  // ctx (HD,Lq) = Vh * P^T (P stored (Lq,Lk) -> BT), f16 only
  gemm(s, 0, 1, 0, Vv16, Sc16, nullptr, Cx16, HDIM, Lq, Lk, Lk, Lk, Lq,
       (ll)EDIM * Lk, (ll)NH * Lq * Lk, (ll)EDIM * Lq,
       NH, (ll)HDIM * Lk, (ll)Lq * Lk, (ll)HDIM * Lq,
       BSZ, nullptr, nullptr, 1.f, nullptr);
  // out-projection (f32 for the LN)
  gemm(s, 0, 0, 0, p.out_w16, Cx16, AtF, nullptr, EDIM, Lq, EDIM, EDIM, Lq, Lq,
       0, (ll)EDIM * Lq, (ll)EDIM * Lq, 1, 0, 0, 0, BSZ, nullptr, p.out_b, 1.f, nullptr);
  const ll tok = (ll)BSZ * Lq;
  ln_residual_kernel<<<(tok + 255) / 256, 256, 0, s>>>(q, q16, AtF, p.ln_g, p.ln_b, Lq);
}

// ======================= entry point ================================
extern "C" void kernel_launch(void* const* d_in, const int* in_sizes, int n_in,
                              void* d_out, int out_size, void* d_ws, size_t ws_size,
                              hipStream_t stream) {
  (void)in_sizes; (void)n_in; (void)out_size; (void)ws_size;
  hipStream_t s = stream;
  auto F = [&](int i) { return (const float*)d_in[i]; };

  const float* rgb_local   = F(0);
  const float* pts_local   = F(1);
  const float* prior_local = F(2);
  const int*   index       = (const int*)d_in[5];
  // params (jax pytree: dict keys sorted, sub-dict keys sorted)
  const float *am0_b = F(6),  *am0_w = F(7),  *am1_b = F(8),  *am1_w = F(9);
  const float *am2_b = F(10), *am2_w = F(11);
  const float *bn1_b = F(12), *bn1_g = F(13), *bn1_m = F(14), *bn1_v = F(15);
  const float *bn2_b = F(16), *bn2_g = F(17), *bn2_m = F(18), *bn2_v = F(19);
  const float *bn3_b = F(20), *bn3_g = F(21), *bn3_m = F(22), *bn3_v = F(23);
  const float *c1_b = F(24), *c1_w = F(25), *c2_b = F(26), *c2_w = F(27);
  const float *c3_b = F(28), *c3_w = F(29);
  const float *d20_b = F(30), *d20_w = F(31), *d21_b = F(32), *d21_w = F(33);
  const float *d22_b = F(34), *d22_w = F(35);
  const float *f30_b = F(52), *f30_w = F(53), *f31_b = F(54), *f31_w = F(55);

  char*  wsb = (char*)d_ws;
  float* out = (float*)d_out;
  int* roffP = (int*)(wsb + B_ROFF);
  int* roff3 = roffP + BSZ;
  auto WF = [&](ll off) { return (float*)(wsb + off); };
  auto WH = [&](ll off) { return (_Float16*)(wsb + off); };

  // ---------------- weight conversion (f32 -> f16, once per call) ------
  _Float16* wb = WH(B_W16);
  ll woff = 0;
  auto cvtw = [&](const float* w, ll n) {
    _Float16* dst = wb + woff;
    cvt_f16_kernel<<<(unsigned)((n + 255) / 256), 256, 0, s>>>(w, dst, n);
    woff += (n + 7) & ~7LL;
    return (const _Float16*)dst;
  };
  const _Float16 *am0_w16 = cvtw(am0_w, 384 * 384), *am1_w16 = cvtw(am1_w, 256 * 384);
  const _Float16 *am2_w16 = cvtw(am2_w, (ll)NCLS * NPR * 256);
  const _Float16 *c1_w16 = cvtw(c1_w, 384 * 384), *c2_w16 = cvtw(c2_w, 256 * 384);
  const _Float16 *c3_w16 = cvtw(c3_w, 128 * 256);
  const _Float16 *d20_w16 = cvtw(d20_w, 64 * 128), *d21_w16 = cvtw(d21_w, 64 * 64);
  const _Float16 *d22_w16 = cvtw(d22_w, 18 * 64);
  const _Float16 *f30_w16 = cvtw(f30_w, 256 * 256), *f31_w16 = cvtw(f31_w, 128 * 256);
  const _Float16 *lin_w16[4];
  for (int i = 0; i < 4; ++i) lin_w16[i] = cvtw(F(57 + 2 * i), 128 * 128);
  const _Float16 *enr_in16[4], *enr_out16[4], *sen_in16[3], *sen_out16[3];
  for (int i = 0; i < 4; ++i) {
    enr_in16[i]  = cvtw(F(37 + 4 * i), 384 * 128);
    enr_out16[i] = cvtw(F(39 + 4 * i), 128 * 128);
  }
  for (int i = 0; i < 3; ++i) {
    sen_in16[i]  = cvtw(F(79 + 4 * i), 384 * 128);
    sen_out16[i] = cvtw(F(81 + 4 * i), 128 * 128);
  }
  cvt_f16_kernel<<<(unsigned)(((ll)BSZ * 128 * NPT + 255) / 256), 256, 0, s>>>(
      pts_local, WH(B_PTSL16), (ll)BSZ * 128 * NPT);
  cvt_f16_kernel<<<(unsigned)(((ll)BSZ * 128 * NPR + 255) / 256), 256, 0, s>>>(
      prior_local, WH(B_PRIORL16), (ll)BSZ * 128 * NPR);

  // ---------------- Phase A: fc3 / lowrank / wpf / globals -------------
  concat2_kernel<<<(BSZ * 256 * NPT + 255) / 256, 256, 0, s>>>(pts_local, rgb_local,
                                                               WH(B_REFCAT16));
  gemm(s, 0, 0, 1, f30_w16, WH(B_REFCAT16), nullptr, WH(B_H1_16), 256, NPT, 256,
       256, NPT, NPT, 0, (ll)256 * NPT, (ll)256 * NPT, 1, 0, 0, 0, BSZ,
       nullptr, f30_b, 1.f, nullptr);
  gemm(s, 0, 0, 0, f31_w16, WH(B_H1_16), WF(B_LOWRF), nullptr, 128, NPT, 256,
       256, NPT, NPT, 0, (ll)256 * NPT, (ll)128 * NPT, 1, 0, 0, 0, BSZ,
       nullptr, f31_b, 1.f, nullptr);
  row_softmax_kernel<<<BSZ * 128, 256, 0, s>>>(WF(B_LOWRF), nullptr, WH(B_LOWR16), NPT);
  row_mean_kernel<<<BSZ * 128, 256, 0, s>>>(rgb_local, WF(B_RGBG), NPT);
  row_mean_kernel<<<BSZ * 128, 256, 0, s>>>(pts_local, WF(B_PTSG), NPT);
  gemm(s, 0, 1, 0, WH(B_PTSL16), WH(B_LOWR16), nullptr, WH(B_WPF16), 128, 128, NPT,
       NPT, NPT, 128, (ll)128 * NPT, (ll)128 * NPT, (ll)128 * 128, 1, 0, 0, 0, BSZ,
       nullptr, nullptr, 1.f, nullptr);

  // ---------------- Phase B: projections + attention stack -------------
  gemm(s, 0, 0, 0, lin_w16[0], WH(B_PRIORL16), WF(B_QPRIORF), WH(B_QPRIOR16), 128, NPR, 128,
       128, NPR, NPR, 0, (ll)128 * NPR, (ll)128 * NPR, 1, 0, 0, 0, BSZ,
       nullptr, F(56), 1.f, nullptr);
  gemm(s, 0, 0, 0, lin_w16[1], WH(B_PTSL16), nullptr, WH(B_KPTS16), 128, NPT, 128,
       128, NPT, NPT, 0, (ll)128 * NPT, (ll)128 * NPT, 1, 0, 0, 0, BSZ,
       nullptr, F(58), 1.f, nullptr);
  gemm(s, 0, 0, 0, lin_w16[2], WH(B_WPF16), WF(B_QPTSF), WH(B_QPTS16), 128, 128, 128,
       128, 128, 128, 0, (ll)128 * 128, (ll)128 * 128, 1, 0, 0, 0, BSZ,
       nullptr, F(60), 1.f, nullptr);

  for (int i = 0; i < 4; ++i) {
    MhaP enr{F(36 + 4 * i), F(38 + 4 * i), F(64 + 2 * i), F(65 + 2 * i),
             enr_in16[i], enr_out16[i]};
    run_mha(s, wsb, WF(B_QPRIORF), WH(B_QPRIOR16), WH(B_QPTS16), NPR, 128, enr);
    if (i < 3) {
      MhaP sen{F(78 + 4 * i), F(80 + 4 * i), F(72 + 2 * i), F(73 + 2 * i),
               sen_in16[i], sen_out16[i]};
      run_mha(s, wsb, WF(B_QPTSF), WH(B_QPTS16), WH(B_KPTS16), 128, NPT, sen);
    }
  }

  // ---------------- Phase C: prior_feat + deformer MLP (BN folded) -----
  gemm(s, 0, 0, 0, lin_w16[3], WH(B_QPRIOR16), WF(B_PFF), nullptr, 128, NPR, 128,
       128, NPR, NPR, 0, (ll)128 * NPR, (ll)128 * NPR, 1, 0, 0, 0, BSZ,
       nullptr, F(62), 1.f, nullptr);
  deform_concat_kernel<<<(BSZ * 384 * NPR + 255) / 256, 256, 0, s>>>(
      WF(B_PFF), WF(B_PTSG), WF(B_RGBG), WH(B_DEF16));
  float *s1 = WF(B_FOLD), *h1v = s1 + 384, *s2 = h1v + 384, *h2v = s2 + 256,
        *s3 = h2v + 256, *h3v = s3 + 128;
  bnfold_kernel<<<2, 256, 0, s>>>(bn1_g, bn1_b, bn1_m, bn1_v, c1_b, s1, h1v, 384);
  bnfold_kernel<<<1, 256, 0, s>>>(bn2_g, bn2_b, bn2_m, bn2_v, c2_b, s2, h2v, 256);
  bnfold_kernel<<<1, 128, 0, s>>>(bn3_g, bn3_b, bn3_m, bn3_v, c3_b, s3, h3v, 128);
  gemm(s, 0, 0, 1, c1_w16, WH(B_DEF16), nullptr, WH(B_X1_16), 384, NPR, 384,
       384, NPR, NPR, 0, (ll)384 * NPR, (ll)384 * NPR, 1, 0, 0, 0, BSZ, s1, h1v, 1.f, nullptr);
  gemm(s, 0, 0, 1, c2_w16, WH(B_X1_16), nullptr, WH(B_X2_16), 256, NPR, 384,
       384, NPR, NPR, 0, (ll)384 * NPR, (ll)256 * NPR, 1, 0, 0, 0, BSZ, s2, h2v, 1.f, nullptr);
  gemm(s, 0, 0, 0, c3_w16, WH(B_X2_16), WF(B_X3F), nullptr, 128, NPR, 256,
       256, NPR, NPR, 0, (ll)256 * NPR, (ll)128 * NPR, 1, 0, 0, 0, BSZ, s3, h3v, 1.f, nullptr);
  addrelu_kernel<<<(unsigned)(((ll)BSZ * 128 * NPR + 255) / 256), 256, 0, s>>>(
      WF(B_PFF), WH(B_PF16), WF(B_X3F), (ll)BSZ * 128 * NPR);
  row_mean_kernel<<<BSZ * 128, 256, 0, s>>>(WF(B_PFF), WF(B_PRIORG), NPR);

  // ---------------- Phase D: attention-map branch (class rows only) ----
  attin_concat_kernel<<<(BSZ * 384 * NPT + 255) / 256, 256, 0, s>>>(
      pts_local, rgb_local, WF(B_PRIORG), WH(B_DEF16));
  gemm(s, 0, 0, 1, am0_w16, WH(B_DEF16), nullptr, WH(B_X1_16), 384, NPT, 384,
       384, NPT, NPT, 0, (ll)384 * NPT, (ll)384 * NPT, 1, 0, 0, 0, BSZ,
       nullptr, am0_b, 1.f, nullptr);
  gemm(s, 0, 0, 1, am1_w16, WH(B_X1_16), nullptr, WH(B_X2_16), 256, NPT, 384,
       384, NPT, NPT, 0, (ll)384 * NPT, (ll)256 * NPT, 1, 0, 0, 0, BSZ,
       nullptr, am1_b, 1.f, nullptr);
  rowoff_kernel<<<1, 32, 0, s>>>(index, roffP, roff3);
  gemm(s, 0, 0, 0, am2_w16, WH(B_X2_16), WF(B_AFSEL), nullptr, NPR, NPT, 256,
       256, NPT, NPT, 0, (ll)256 * NPT, (ll)NPR * NPT, 1, 0, 0, 0, BSZ,
       nullptr, am2_b, 1.f, roffP);
  transpose_kernel<<<dim3(NPT / 32, NPR / 32, BSZ), 256, 0, s>>>(
      WF(B_AFSEL), out + O_OUT_ATT, NPR, NPT);
  row_softmax_kernel<<<BSZ * NPT, 256, 0, s>>>(out + O_OUT_ATT, WF(B_SOFTAF),
                                               WH(B_SOFTA16), NPR);

  // ---------------- Phase E: outputs -----------------------------------
  gemm(s, 0, 1, 0, WH(B_PF16), WH(B_SOFTA16), out + O_OUT_NP, nullptr, 128, NPT, NPR,
       NPR, NPR, NPT, (ll)128 * NPR, (ll)NPT * NPR, (ll)128 * NPT,
       1, 0, 0, 0, BSZ, nullptr, nullptr, 1.f, nullptr);
  gemm(s, 0, 0, 1, d20_w16, WH(B_PF16), nullptr, WH(B_QV1_16), 64, NPR, 128,
       128, NPR, NPR, 0, (ll)128 * NPR, (ll)64 * NPR, 1, 0, 0, 0, BSZ,
       nullptr, d20_b, 1.f, nullptr);
  gemm(s, 0, 0, 1, d21_w16, WH(B_QV1_16), nullptr, WH(B_QV2_16), 64, NPR, 64,
       64, NPR, NPR, 0, (ll)64 * NPR, (ll)64 * NPR, 1, 0, 0, 0, BSZ,
       nullptr, d21_b, 1.f, nullptr);
  gemm(s, 0, 0, 0, d22_w16, WH(B_QV2_16), WF(B_QV3F), nullptr, 3, NPR, 64,
       64, NPR, NPR, 0, (ll)64 * NPR, (ll)3 * NPR, 1, 0, 0, 0, BSZ,
       nullptr, d22_b, 1.f, roff3);
  qv_out_kernel<<<(BSZ * NPR + 255) / 256, 256, 0, s>>>(WF(B_QV3F), out + O_OUT_QV);
  qo_kernel<<<(BSZ * NPT + 255) / 256, 256, 0, s>>>(out + O_OUT_QV, WF(B_SOFTAF),
                                                    out + O_OUT_QO);
}
